// Discriminator_55800215109843
// MI455X (gfx1250) — matrix operand order repaired and checked by reading, not compiled, not run on hardware
//
#include <hip/hip_runtime.h>
#include <hip/hip_bf16.h>

typedef __attribute__((ext_vector_type(16))) _Float16 v16h;
typedef __attribute__((ext_vector_type(8)))  _Float16 h8;
typedef __attribute__((ext_vector_type(8)))  float    v8f;
typedef __attribute__((ext_vector_type(4)))  float    v4f;

#define LIF_BETA 0.95f
#define LIF_THR  1.0f
#define LIF_T    50
#define BN_EPS   0.8f

// ---------------------------------------------------------------------------
// Weight repack: w[Cout,Cin,3,3] f32  ->  wp[Cout, KTpad] f16, K = tap*CIN+ci,
// zero-padded to KTpad = ceil(9*CIN/32)*32.  A-fragments then load as two
// contiguous b128s with no bounds checks.
// ---------------------------------------------------------------------------
__global__ __launch_bounds__(256) void pack_weights_kernel(
    const float* __restrict__ w, _Float16* __restrict__ wp,
    int CIN, int COUT, int KTpad)
{
    int i = blockIdx.x * 256 + threadIdx.x;
    if (i >= COUT * KTpad) return;
    int co = i / KTpad, k = i % KTpad;
    int KT = CIN * 9;
    _Float16 v = (_Float16)0.f;
    if (k < KT) {
        int tap = k / CIN, ci = k % CIN;
        v = (_Float16)w[(co * CIN + ci) * 9 + tap];
    }
    wp[i] = v;
}

// ---------------------------------------------------------------------------
// Layer 1: direct conv 3->16, k3 s2 p1, LeakyReLU(0.2).
// f32 NCHW img -> f16 channels-last out1 [B,64,64,16] (two b128 stores/pixel).
// ---------------------------------------------------------------------------
__global__ __launch_bounds__(256) void conv1_lrelu_kernel(
    const float* __restrict__ img, const float* __restrict__ w1,
    const float* __restrict__ b1, _Float16* __restrict__ out1)
{
    __shared__ float ws[16 * 27];
    __shared__ float bs[16];
    for (int i = threadIdx.x; i < 16 * 27; i += 256) ws[i] = w1[i];
    if (threadIdx.x < 16) bs[threadIdx.x] = b1[threadIdx.x];
    __syncthreads();

    int t  = blockIdx.x * 256 + threadIdx.x;   // 0 .. 128*64*64-1
    int ow = t & 63;
    int oh = (t >> 6) & 63;
    int b  = t >> 12;

    float win[27];
#pragma unroll
    for (int ci = 0; ci < 3; ++ci)
#pragma unroll
        for (int kh = 0; kh < 3; ++kh)
#pragma unroll
            for (int kw = 0; kw < 3; ++kw) {
                int ih = oh * 2 - 1 + kh, iw = ow * 2 - 1 + kw;
                bool inb = (ih >= 0) & (ih < 128) & (iw >= 0) & (iw < 128);
                int ihc = ih < 0 ? 0 : (ih > 127 ? 127 : ih);
                int iwc = iw < 0 ? 0 : (iw > 127 ? 127 : iw);
                float v = img[((b * 3 + ci) * 128 + ihc) * 128 + iwc];
                win[ci * 9 + kh * 3 + kw] = inb ? v : 0.f;
            }

    h8 p0, p1;
#pragma unroll
    for (int co = 0; co < 16; ++co) {
        float s = bs[co];
#pragma unroll
        for (int k = 0; k < 27; ++k) s = fmaf(win[k], ws[co * 27 + k], s);
        s = (s >= 0.f) ? s : 0.2f * s;
        if (co < 8) p0[co] = (_Float16)s; else p1[co - 8] = (_Float16)s;
    }
    size_t base = (((size_t)b * 64 + oh) * 64 + ow) * 16;
    *(h8*)(out1 + base)     = p0;
    *(h8*)(out1 + base + 8) = p1;
}

// ---------------------------------------------------------------------------
// Layers 2-4: implicit-GEMM conv (k3 s2 p1) + bias + LeakyReLU + BatchNorm
// via v_wmma_f32_16x16x32_f16.  One wave32 owns a 16(Cout) x 64(N) tile.
// Activations channels-last [B,H,W,C] f16; weights pre-packed [COUT,KTpad].
// All loads are unconditional b128s from clamped addresses + value selects.
// A-frag: elem e -> K = hi*16 + e (two contiguous h8).
// B-frag: elem e -> K = hi*16 + e; a 16-K run = one tap, 16 contiguous ci.
// ---------------------------------------------------------------------------
template <int CIN, int COUT, int IH, int OH, bool OUT_F32>
__global__ __launch_bounds__(128) void conv_bn_wmma_kernel(
    const _Float16* __restrict__ in, const _Float16* __restrict__ wp,
    const float* __restrict__ bias,
    const float* __restrict__ gamma, const float* __restrict__ beta,
    const float* __restrict__ rmean, const float* __restrict__ rvar,
    _Float16* __restrict__ out_h, float* __restrict__ out_f)
{
    constexpr int IW = IH, OW = OH;
    constexpr int KT = CIN * 9;
    constexpr int CHUNKS = (KT + 31) / 32;
    constexpr int KTpad = CHUNKS * 32;
    constexpr int CO_TILES = COUT / 16;
    constexpr int NTOT = 128 * OH * OW;

    const int lane = threadIdx.x & 31;
    const int col  = lane & 15;
    const int hi   = lane >> 4;
    const int wave = blockIdx.x * 4 + (threadIdx.x >> 5);
    const int co_tile = wave % CO_TILES;
    const int n_base  = (wave / CO_TILES) * 64;
    if (n_base >= NTOT) return;   // uniform per-wave: EXEC stays all-ones

    const _Float16* arow = wp + (size_t)(co_tile * 16 + col) * KTpad;

    int nb[4], noh[4], now[4];
#pragma unroll
    for (int s = 0; s < 4; ++s) {
        int n = n_base + s * 16 + col;
        nb[s]  = n / (OH * OW);
        int hw = n % (OH * OW);
        noh[s] = hw / OW;
        now[s] = hw % OW;
    }

    const v8f vzero = {0.f, 0.f, 0.f, 0.f, 0.f, 0.f, 0.f, 0.f};
    const h8  hzero = {(_Float16)0.f, (_Float16)0.f, (_Float16)0.f, (_Float16)0.f,
                       (_Float16)0.f, (_Float16)0.f, (_Float16)0.f, (_Float16)0.f};
    v8f acc[4];
#pragma unroll
    for (int s = 0; s < 4; ++s) acc[s] = vzero;

#pragma unroll
    for (int kc = 0; kc < CHUNKS; ++kc) {
        const int k0 = kc * 32;

        // ---- A fragment: two contiguous 16B loads from packed weights ----
        h8 alo = *(const h8*)(arow + k0 + hi * 16);
        h8 ahi = *(const h8*)(arow + k0 + hi * 16 + 8);
        v16h afrag = __builtin_shufflevector(alo, ahi,
            0, 1, 2, 3, 4, 5, 6, 7, 8, 9, 10, 11, 12, 13, 14, 15);

        // This lane's 16-K run: one tap, 16 contiguous input channels.
        const int kb   = k0 + hi * 16;
        const int tap  = kb / CIN;          // constant-folds except layer-2 tail
        const bool kok = (tap < 9);
        const int tapc = kok ? tap : 8;
        const int ci0  = kb % CIN;
        const int kh = tapc / 3, kw = tapc % 3;

#pragma unroll
        for (int s = 0; s < 4; ++s) {
            int ih = noh[s] * 2 - 1 + kh;
            int iw = now[s] * 2 - 1 + kw;
            bool inb = kok & (ih >= 0) & (ih < IH) & (iw >= 0) & (iw < IW);
            int ihc = ih < 0 ? 0 : (ih > IH - 1 ? IH - 1 : ih);
            int iwc = iw < 0 ? 0 : (iw > IW - 1 ? IW - 1 : iw);
            const _Float16* bp =
                in + (((size_t)nb[s] * IH + ihc) * IW + iwc) * CIN + ci0;
            h8 blo = *(const h8*)bp;
            h8 bhi = *(const h8*)(bp + 8);
            blo = inb ? blo : hzero;        // v_cndmask, no exec branching
            bhi = inb ? bhi : hzero;
            v16h bfrag = __builtin_shufflevector(blo, bhi,
                0, 1, 2, 3, 4, 5, 6, 7, 8, 9, 10, 11, 12, 13, 14, 15);
            acc[s] = __builtin_amdgcn_wmma_f32_16x16x32_f16(
                false, afrag, false, bfrag, (short)0, acc[s], false, false);
        }
    }

    // ---- Epilogue: bias + LeakyReLU + BatchNorm; contiguous stores ----
    float bsv[8], sc[8], sh[8];
#pragma unroll
    for (int r = 0; r < 8; ++r) {
        int co = co_tile * 16 + hi * 8 + r;
        bsv[r] = bias[co];
        float scale = gamma[co] * rsqrtf(rvar[co] + BN_EPS);
        sc[r] = scale;
        sh[r] = beta[co] - rmean[co] * scale;
    }
#pragma unroll
    for (int s = 0; s < 4; ++s) {
        float xr[8];
        h8 pack;
#pragma unroll
        for (int r = 0; r < 8; ++r) {
            float x = acc[s][r] + bsv[r];
            x = (x >= 0.f) ? x : 0.2f * x;
            x = x * sc[r] + sh[r];
            xr[r] = x;
            pack[r] = (_Float16)x;
        }
        size_t base = (((size_t)nb[s] * OH + noh[s]) * OW + now[s]) * COUT
                      + co_tile * 16 + hi * 8;
        if constexpr (OUT_F32) {
            v4f lo = {xr[0], xr[1], xr[2], xr[3]};
            v4f hi4 = {xr[4], xr[5], xr[6], xr[7]};
            *(v4f*)(out_f + base)     = lo;
            *(v4f*)(out_f + base + 4) = hi4;
        } else {
            *(h8*)(out_h + base) = pack;
        }
    }
}

// ---------------------------------------------------------------------------
// LIF (constant-input, subtract-reset) fused with final linear + sigmoid.
// conv4 out is channels-last [B,8,8,128] f32; each thread owns wl columns
// idx = c*64+hw (coalesced wl reads, which dominate traffic).
// ---------------------------------------------------------------------------
__global__ __launch_bounds__(256) void lif_linear_sigmoid_kernel(
    const float* __restrict__ out4, const float* __restrict__ wl,
    const float* __restrict__ bl, float* __restrict__ out)
{
    const int b   = blockIdx.x;
    const int tid = threadIdx.x;
    float acc = 0.f;

#pragma unroll 1
    for (int j = 0; j < 32; ++j) {
        int idx = j * 256 + tid;                 // = c*64 + hw (wl column)
        int c   = idx >> 6;
        int hw  = idx & 63;
        float x = out4[b * 8192 + hw * 128 + c]; // channels-last gather (rare)
        const float* wp = wl + idx;
        float mem = 0.f;
#pragma unroll 1
        for (int t = 0; t < LIF_T; ++t) {
            float reset = (mem > LIF_THR) ? LIF_THR : 0.f;   // pre-update mem
            mem = fmaf(LIF_BETA, mem, x) - reset;
            float wv = wp[t * 8192];                         // coalesced per t
            acc += (mem > LIF_THR) ? wv : 0.f;
        }
    }

    __shared__ float red[256];
    red[tid] = acc;
    __syncthreads();
#pragma unroll
    for (int s = 128; s > 0; s >>= 1) {
        if (tid < s) red[tid] += red[tid + s];
        __syncthreads();
    }
    if (tid == 0) {
        float y = red[0] + bl[0];
        out[b] = 1.f / (1.f + expf(-y));
    }
}

// ---------------------------------------------------------------------------
extern "C" void kernel_launch(void* const* d_in, const int* in_sizes, int n_in,
                              void* d_out, int out_size, void* d_ws, size_t ws_size,
                              hipStream_t stream)
{
    const float* img = (const float*)d_in[0];
    const float* w1  = (const float*)d_in[1];
    const float* b1  = (const float*)d_in[2];
    const float* w2  = (const float*)d_in[3];
    const float* b2  = (const float*)d_in[4];
    const float* w3  = (const float*)d_in[5];
    const float* b3  = (const float*)d_in[6];
    const float* w4  = (const float*)d_in[7];
    const float* b4  = (const float*)d_in[8];
    const float* g2  = (const float*)d_in[9];
    const float* bb2 = (const float*)d_in[10];
    const float* rm2 = (const float*)d_in[11];
    const float* rv2 = (const float*)d_in[12];
    const float* g3  = (const float*)d_in[13];
    const float* bb3 = (const float*)d_in[14];
    const float* rm3 = (const float*)d_in[15];
    const float* rv3 = (const float*)d_in[16];
    const float* g4  = (const float*)d_in[17];
    const float* bb4 = (const float*)d_in[18];
    const float* rm4 = (const float*)d_in[19];
    const float* rv4 = (const float*)d_in[20];
    const float* wl  = (const float*)d_in[21];
    const float* bl  = (const float*)d_in[22];
    float* out = (float*)d_out;

    // Workspace layout (halves unless noted); all 16B aligned.
    _Float16* out1 = (_Float16*)d_ws;                            // 128*64*64*16
    _Float16* out2 = out1 + (size_t)128 * 64 * 64 * 16;          // 128*32*32*32
    _Float16* out3 = out2 + (size_t)128 * 32 * 32 * 32;          // 128*16*16*64
    float*    out4 = (float*)(out3 + (size_t)128 * 16 * 16 * 64);// 128*8*8*128 f32
    _Float16* wp2  = (_Float16*)(out4 + (size_t)128 * 8 * 8 * 128);
    _Float16* wp3  = wp2 + (size_t)32 * 160;    // KTpad2 = 160
    _Float16* wp4  = wp3 + (size_t)64 * 288;    // KTpad3 = 288; KTpad4 = 576

    // Weight repacks (independent of activations).
    pack_weights_kernel<<<(32 * 160 + 255) / 256, 256, 0, stream>>>(w2, wp2, 16, 32, 160);
    pack_weights_kernel<<<(64 * 288 + 255) / 256, 256, 0, stream>>>(w3, wp3, 32, 64, 288);
    pack_weights_kernel<<<(128 * 576 + 255) / 256, 256, 0, stream>>>(w4, wp4, 64, 128, 576);

    // L1: [128,3,128,128] -> [128,64,64,16]
    conv1_lrelu_kernel<<<2048, 256, 0, stream>>>(img, w1, b1, out1);

    // L2: Cin=16 Cout=32 64->32 : waves = 2 * (131072/64) = 4096 -> 1024 blocks
    conv_bn_wmma_kernel<16, 32, 64, 32, false><<<1024, 128, 0, stream>>>(
        out1, wp2, b2, g2, bb2, rm2, rv2, out2, nullptr);

    // L3: Cin=32 Cout=64 32->16 : waves = 4 * 512 = 2048 -> 512 blocks
    conv_bn_wmma_kernel<32, 64, 32, 16, false><<<512, 128, 0, stream>>>(
        out2, wp3, b3, g3, bb3, rm3, rv3, out3, nullptr);

    // L4: Cin=64 Cout=128 16->8 : waves = 8 * 128 = 1024 -> 256 blocks
    conv_bn_wmma_kernel<64, 128, 16, 8, true><<<256, 128, 0, stream>>>(
        out3, wp4, b4, g4, bb4, rm4, rv4, nullptr, out4);

    // LIF + linear + sigmoid -> [128,1]
    lif_linear_sigmoid_kernel<<<128, 256, 0, stream>>>(out4, wl, bl, out);
}
